// RBFKernel_59605556134141
// MI455X (gfx1250) — matrix operand order repaired and checked by reading, not compiled, not run on hardware
//
#include <hip/hip_runtime.h>
#include <hip/hip_bf16.h>

// ---------------------------------------------------------------------------
// RBF kernel: out[b,m,n] = exp(-(||X[b,m]||^2 + ||Y[b,n]||^2 - 2*X[b,m].Y[b,n])/2)
// B=4, N=M=4096, D=256.
//   Pass 1: f32 row norms + f32->bf16 conversion of X,Y into workspace.
//   Pass 2: bf16 GEMM via v_wmma_f32_16x16x32_bf16, tiles staged into LDS by
//           the Tensor Data Mover (tensor_load_to_lds + s_wait_tensorcnt),
//           double-buffered; f32 norm terms folded in at the exp epilogue.
// ---------------------------------------------------------------------------

typedef __attribute__((ext_vector_type(8)))  float        v8f;
typedef __attribute__((ext_vector_type(16))) __bf16       v16bf;
typedef __attribute__((ext_vector_type(4)))  __bf16       bf16x4;
typedef __attribute__((ext_vector_type(4)))  unsigned int v4u;
typedef __attribute__((ext_vector_type(8)))  int          v8i;
typedef __attribute__((ext_vector_type(4)))  int          v4i;

#define BATCH 4
#define NDIM  4096
#define KDIM  256
#define TILE  128      // output tile per block (128x128)
#define KT    32       // K per LDS stage (one WMMA K-depth)
#define NSTAGE (KDIM / KT)
#define LDS_STRIDE 40  // 32 data + 8 pad bf16 (64B data + 16B pad per row)

union FragAB {
    uint4 u[2];   // 2 x 16B
    v16bf v;      // 16 bf16 = 32 B = 8 VGPRs
};

// ---------------------------------------------------------------------------
// Pass 1: one wave32 per row. Lane handles 8 contiguous elements:
// accumulate |x|^2 in f32, emit bf16 copy; wave-reduce norm via shfl_xor.
// ---------------------------------------------------------------------------
__global__ void __launch_bounds__(256)
rbf_prep_kernel(const float* __restrict__ X, const float* __restrict__ Y,
                float* __restrict__ x2, float* __restrict__ y2,
                __bf16* __restrict__ Xbf, __bf16* __restrict__ Ybf) {
    const int lane = threadIdx.x & 31;
    const int warp = threadIdx.x >> 5;
    const int row  = blockIdx.x * 8 + warp;       // 8 rows per 256-thread block

    const float4* px = (const float4*)(X + (size_t)row * KDIM) + lane * 2;
    const float4* py = (const float4*)(Y + (size_t)row * KDIM) + lane * 2;
    bf16x4* qx = (bf16x4*)(Xbf + (size_t)row * KDIM) + lane * 2;
    bf16x4* qy = (bf16x4*)(Ybf + (size_t)row * KDIM) + lane * 2;

    float4 a0 = px[0], a1 = px[1];
    float4 b0 = py[0], b1 = py[1];

    float sx = a0.x*a0.x + a0.y*a0.y + a0.z*a0.z + a0.w*a0.w
             + a1.x*a1.x + a1.y*a1.y + a1.z*a1.z + a1.w*a1.w;
    float sy = b0.x*b0.x + b0.y*b0.y + b0.z*b0.z + b0.w*b0.w
             + b1.x*b1.x + b1.y*b1.y + b1.z*b1.z + b1.w*b1.w;

    bf16x4 xa0 = { (__bf16)a0.x, (__bf16)a0.y, (__bf16)a0.z, (__bf16)a0.w };
    bf16x4 xa1 = { (__bf16)a1.x, (__bf16)a1.y, (__bf16)a1.z, (__bf16)a1.w };
    bf16x4 yb0 = { (__bf16)b0.x, (__bf16)b0.y, (__bf16)b0.z, (__bf16)b0.w };
    bf16x4 yb1 = { (__bf16)b1.x, (__bf16)b1.y, (__bf16)b1.z, (__bf16)b1.w };
    qx[0] = xa0; qx[1] = xa1;
    qy[0] = yb0; qy[1] = yb1;

#pragma unroll
    for (int m = 16; m > 0; m >>= 1) {
        sx += __shfl_xor(sx, m, 32);
        sy += __shfl_xor(sy, m, 32);
    }
    if (lane == 0) { x2[row] = sx; y2[row] = sy; }
}

// ---------------------------------------------------------------------------
// TDM: load a 128-row x 32-col bf16 tile (row stride KDIM elements in memory)
// into LDS at lds_byte, inserting 16B pad after each 64B row (-> 80B rows,
// matching LDS_STRIDE=40 bf16).  D# per CDNA5 ISA.
// ---------------------------------------------------------------------------
__device__ __forceinline__ void tdm_load_tile(const __bf16* tile, unsigned lds_byte) {
    unsigned long long ga = (unsigned long long)(uintptr_t)tile;

    v4u g0;
    g0.x = 1u;                                          // count=1 (valid D#)
    g0.y = lds_byte;                                    // lds_addr
    g0.z = (unsigned)ga;                                // global_addr[31:0]
    g0.w = (unsigned)((ga >> 32) & 0x1FFFFFFu)          // global_addr[56:32]
         | (2u << 30);                                  // type=2 ("image")

    v8i g1;
    g1[0] = (int)((1u << 16)      // data_size=1 -> 2-byte elements
                | (1u << 20)      // pad_enable
                | (3u << 22)      // pad_interval: 16 DWORDs (=64B row)
                | (3u << 25));    // pad_amount:   4 DWORDs (=16B pad)
    g1[1] = (int)((unsigned)KDIM << 16);   // tensor_dim0[15:0] = 256
    g1[2] = (int)((unsigned)NDIM << 16);   // tensor_dim1[15:0] = 4096
    g1[3] = (int)((unsigned)KT   << 16);   // tile_dim0 = 32
    g1[4] = TILE;                          // tile_dim1 = 128 (tile_dim2 = 0)
    g1[5] = KDIM;                          // tensor_dim0_stride[31:0] = 256
    g1[6] = 0;
    g1[7] = 0;

    v4i z4 = { 0, 0, 0, 0 };
#if __clang_major__ >= 23
    v8i z8 = { 0, 0, 0, 0, 0, 0, 0, 0 };
    __builtin_amdgcn_tensor_load_to_lds(g0, g1, z4, z4, z8, 0);
#else
    __builtin_amdgcn_tensor_load_to_lds(g0, g1, z4, z4, 0);
#endif
}

// ---------------------------------------------------------------------------
// Pass 2: tiled GEMM + RBF epilogue.
// grid = (N/TILE, N/TILE, BATCH), block = 256 = 8 wave32 waves (4x2 grid),
// each wave owns 32(M) x 64(N) = 2x4 WMMA tiles. LDS double-buffered; wave 0
// drives the TDM, compute of stage k overlaps the DMA of stage k+1.
// ---------------------------------------------------------------------------
__global__ void __launch_bounds__(256)
rbf_gemm_kernel(const __bf16* __restrict__ Xbf,
                const __bf16* __restrict__ Ybf,
                const float* __restrict__ x2,
                const float* __restrict__ y2,
                float* __restrict__ out) {
    __shared__ __bf16 Xs[2][TILE * LDS_STRIDE];
    __shared__ __bf16 Ys[2][TILE * LDS_STRIDE];

    const int t    = threadIdx.x;
    const int lane = t & 31;
    const int wave = t >> 5;          // 0..7
    const int wm   = wave >> 1;       // 0..3 : 32-row slice of M
    const int wn   = wave & 1;        // 0..1 : 64-col slice of N

    const int laneLo = lane & 15;
    const int hi     = lane >> 4;
    const int hi8    = hi * 8;

    const int b      = blockIdx.z;
    const int tileM0 = blockIdx.y * TILE;
    const int tileN0 = blockIdx.x * TILE;

    const __bf16* Xtile0 = Xbf + ((size_t)(b * NDIM + tileM0)) * KDIM;
    const __bf16* Ytile0 = Ybf + ((size_t)(b * NDIM + tileN0)) * KDIM;

    v8f acc[2][4] = {};

    // Prologue: stage 0 into buffer 0.
    if (wave == 0) {
        tdm_load_tile(Xtile0, (unsigned)(uintptr_t)&Xs[0][0]);
        tdm_load_tile(Ytile0, (unsigned)(uintptr_t)&Ys[0][0]);
    }

    for (int kt = 0; kt < NSTAGE; ++kt) {
        const int cur = kt & 1;

        if (wave == 0) __builtin_amdgcn_s_wait_tensorcnt(0);
        __syncthreads();   // buf[cur] visible to all waves

        // Kick off stage kt+1 into the other buffer; overlaps compute below.
        if (wave == 0 && kt + 1 < NSTAGE) {
            const int k1 = (kt + 1) * KT;
            tdm_load_tile(Xtile0 + k1, (unsigned)(uintptr_t)&Xs[cur ^ 1][0]);
            tdm_load_tile(Ytile0 + k1, (unsigned)(uintptr_t)&Ys[cur ^ 1][0]);
        }

        // --- A fragments: 16x32 bf16. Lanes 0-15: M=lane, K {0..15};
        //     lanes 16-31: M=lane-16, K {16..31}.
        FragAB A[2];
#pragma unroll
        for (int tm = 0; tm < 2; ++tm) {
            const __bf16* base = &Xs[cur][(wm * 32 + tm * 16 + laneLo) * LDS_STRIDE + hi * 16];
            A[tm].u[0] = *(const uint4*)(base);
            A[tm].u[1] = *(const uint4*)(base + 8);
        }
        // --- B fragments: 32x16 bf16. Lane = column N, contiguous 16-K chunk
        //     at K = hi*16.
        FragAB Bf[4];
#pragma unroll
        for (int tn = 0; tn < 4; ++tn) {
            const __bf16* base = &Ys[cur][(wn * 64 + tn * 16 + laneLo) * LDS_STRIDE + hi * 16];
            Bf[tn].u[0] = *(const uint4*)(base);
            Bf[tn].u[1] = *(const uint4*)(base + 8);
        }

#pragma unroll
        for (int tm = 0; tm < 2; ++tm)
#pragma unroll
            for (int tn = 0; tn < 4; ++tn)
                acc[tm][tn] = __builtin_amdgcn_wmma_f32_16x16x32_bf16(
                    false, A[tm].v, false, Bf[tn].v,
                    (short)0, acc[tm][tn], false, false);

        __syncthreads();   // all waves done reading buf[cur]
    }

    // ------------------------- epilogue -------------------------
    // C/D layout: VGPR r, lanes 0-15 -> (M=r, N=lane); lanes 16-31 -> M=r+8.
    const float* x2b = x2 + b * NDIM;
    const float* y2b = y2 + b * NDIM;
    float* outb = out + (size_t)b * NDIM * NDIM;

#pragma unroll
    for (int tm = 0; tm < 2; ++tm) {
        const int mBase = tileM0 + wm * 32 + tm * 16 + hi8;
#pragma unroll
        for (int tn = 0; tn < 4; ++tn) {
            const int n    = tileN0 + wn * 64 + tn * 16 + laneLo;
            const float yn = y2b[n];
#pragma unroll
            for (int r = 0; r < 8; ++r) {
                const int m = mBase + r;
                float d = x2b[m] + yn - 2.0f * acc[tm][tn][r];
                d = fmaxf(d, 0.0f);
                outb[(size_t)m * NDIM + n] = __expf(-0.5f * d);
            }
        }
    }
}

// ---------------------------------------------------------------------------
extern "C" void kernel_launch(void* const* d_in, const int* in_sizes, int n_in,
                              void* d_out, int out_size, void* d_ws, size_t ws_size,
                              hipStream_t stream) {
    const float* X = (const float*)d_in[0];
    const float* Y = (const float*)d_in[1];
    float* out = (float*)d_out;

    // Workspace layout: x2 | y2 | Xbf | Ybf  (64KB + 64KB + 8MB + 8MB)
    float*  x2  = (float*)d_ws;
    float*  y2  = x2 + BATCH * NDIM;
    __bf16* Xbf = (__bf16*)(y2 + BATCH * NDIM);
    __bf16* Ybf = Xbf + (size_t)BATCH * NDIM * KDIM;

    {
        dim3 block(256);
        dim3 grid(BATCH * NDIM / 8);   // 8 rows (waves) per block
        rbf_prep_kernel<<<grid, block, 0, stream>>>(X, Y, x2, y2, Xbf, Ybf);
    }
    {
        dim3 block(256);
        dim3 grid(NDIM / TILE, NDIM / TILE, BATCH);
        rbf_gemm_kernel<<<grid, block, 0, stream>>>(Xbf, Ybf, x2, y2, out);
    }
}
